// Encoder_79396765434009
// MI455X (gfx1250) — matrix operand order repaired and checked by reading, not compiled, not run on hardware
//
#include <hip/hip_runtime.h>
#include <hip/hip_bf16.h>

typedef __attribute__((ext_vector_type(16))) __bf16 v16bf;
typedef __attribute__((ext_vector_type(8)))  __bf16 v8bf;
typedef __attribute__((ext_vector_type(8)))  float  v8f;
typedef __attribute__((ext_vector_type(4)))  float  fvec4;

#define NNODES 16384
#define DIM    128
#define TILE_M 128
#define TILE_K 64
#define NCHUNK (NNODES / TILE_K)

union BF16x16 { v16bf v; v8bf h[2]; };

// Computes Hout = relu((A @ Hin) @ W) for a 128-row strip per block.
// A: NNODES x NNODES fp32 (streamed non-temporally), Hin: NNODES x DIM fp32
// (L2-resident), W: DIM x DIM fp32 ([in][out]), Hout: NNODES x DIM fp32.
__global__ __launch_bounds__(256) void gcn_layer_kernel(
    const float* __restrict__ A, const float* __restrict__ Hin,
    const float* __restrict__ W, float* __restrict__ Hout)
{
    __shared__ __align__(32) __bf16 Asub[2][TILE_M][TILE_K]; // 32 KB double-buffered
    __shared__ __align__(32) __bf16 Htl[2][DIM][TILE_K];     // 32 KB double-buffered
    __shared__ __align__(32) __bf16 Msub[TILE_M][DIM];       // 32 KB (A@Hin tile, bf16)
    __shared__ __align__(32) __bf16 Wt[DIM][DIM];            // 32 KB (W transposed)

    const int tid     = threadIdx.x;
    const int lane    = tid & 31;
    const int wave    = tid >> 5;
    const int lmod    = lane & 15;     // N (B/C) or M (A) within a 16-tile
    const int halfSel = lane >> 4;     // half-wave select per ISA layout
    const size_t rowBase = (size_t)blockIdx.x * TILE_M;
    const int waveRow = wave * 16;

    // Stage W transposed as bf16 (read only in the epilogue).
    for (int i = 0; i < 64; ++i) {
        int idx = i * 256 + tid;              // 0..16383
        int k = idx >> 7, n = idx & 127;      // W is [in=k][out=n]
        Wt[n][k] = (__bf16)W[idx];
    }

    fvec4 aReg[8];   // staged A chunk (global -> regs -> LDS)
    fvec4 hReg[8];   // staged Hin chunk

    auto loadTiles = [&](int k0) {
        #pragma unroll
        for (int i = 0; i < 8; ++i) {
            int f = i * 256 + tid;            // float4 index, 0..2047
            int r = f >> 4;                   // 16 float4 per 64-wide row
            int c = (f & 15) << 2;
            // A is single-use streaming data: non-temporal, don't pollute L2.
            aReg[i] = __builtin_nontemporal_load(
                (const fvec4*)(A + (rowBase + r) * (size_t)NNODES + k0 + c));
        }
        #pragma unroll
        for (int i = 0; i < 8; ++i) {
            int f = i * 256 + tid;            // 0..2047
            int kk = f >> 5;                  // 32 float4 per 128-wide row
            int c = (f & 31) << 2;
            // Hin fits in L2 and is re-read by every block: keep default RT.
            hReg[i] = *(const fvec4*)(Hin + (size_t)(k0 + kk) * DIM + c);
        }
    };

    auto storeTiles = [&](int b) {
        #pragma unroll
        for (int i = 0; i < 8; ++i) {
            int f = i * 256 + tid;
            int r = f >> 4;
            int c = (f & 15) << 2;
            Asub[b][r][c + 0] = (__bf16)aReg[i][0];
            Asub[b][r][c + 1] = (__bf16)aReg[i][1];
            Asub[b][r][c + 2] = (__bf16)aReg[i][2];
            Asub[b][r][c + 3] = (__bf16)aReg[i][3];
        }
        #pragma unroll
        for (int i = 0; i < 8; ++i) {
            int f = i * 256 + tid;
            int kk = f >> 5;
            int c = (f & 31) << 2;
            // Transposed store so B-fragments are contiguous LDS rows.
            Htl[b][c + 0][kk] = (__bf16)hReg[i][0];
            Htl[b][c + 1][kk] = (__bf16)hReg[i][1];
            Htl[b][c + 2][kk] = (__bf16)hReg[i][2];
            Htl[b][c + 3][kk] = (__bf16)hReg[i][3];
        }
    };

    v8f acc[8] = {};   // 8 N-tiles x 8 f32 accum VGPRs

    auto compute = [&](int b) {
        #pragma unroll
        for (int ks = 0; ks < 2; ++ks) {
            BF16x16 afrag;
            // A-frag (16x32 bf16): lane m = lmod; 16 contiguous K at halfSel*16
            const __bf16* ap = &Asub[b][waveRow + lmod][ks * 32 + halfSel * 16];
            afrag.h[0] = *(const v8bf*)ap;
            afrag.h[1] = *(const v8bf*)(ap + 8);
            #pragma unroll
            for (int nt = 0; nt < 8; ++nt) {
                // B-frag (32x16 bf16): lane n = lmod; 16 contiguous K at halfSel*16
                v16bf bb = *(const v16bf*)(&Htl[b][nt * 16 + lmod][ks * 32 + halfSel * 16]);
                acc[nt] = __builtin_amdgcn_wmma_f32_16x16x32_bf16(
                    false, afrag.v, false, bb, (short)0, acc[nt], false, false);
            }
        }
    };

    // ---- software-pipelined main loop: overlap global loads with WMMA ----
    loadTiles(0);
    storeTiles(0);
    __syncthreads();

    for (int c = 0; c < NCHUNK; ++c) {
        const int cur = c & 1;
        if (c + 1 < NCHUNK) loadTiles((c + 1) * TILE_K);  // global loads in flight
        compute(cur);                                      // WMMA on current buffer
        if (c + 1 < NCHUNK) storeTiles(cur ^ 1);           // fill the other buffer
        __syncthreads();
    }

    // Spill (A@Hin) tile to LDS as bf16 per C/D layout: M = v + 8*halfSel, N = lmod.
    #pragma unroll
    for (int nt = 0; nt < 8; ++nt)
        #pragma unroll
        for (int vi = 0; vi < 8; ++vi)
            Msub[waveRow + halfSel * 8 + vi][nt * 16 + lmod] = (__bf16)acc[nt][vi];
    __syncthreads();

    // Epilogue GEMM: (128x128 M-tile) x (128x128 W), fused ReLU, fp32 store.
    v8f acc2[8] = {};
    #pragma unroll
    for (int ks = 0; ks < 4; ++ks) {
        BF16x16 afrag;
        const __bf16* ap = &Msub[waveRow + lmod][ks * 32 + halfSel * 16];
        afrag.h[0] = *(const v8bf*)ap;
        afrag.h[1] = *(const v8bf*)(ap + 8);
        #pragma unroll
        for (int nt = 0; nt < 8; ++nt) {
            v16bf bb = *(const v16bf*)(&Wt[nt * 16 + lmod][ks * 32 + halfSel * 16]);
            acc2[nt] = __builtin_amdgcn_wmma_f32_16x16x32_bf16(
                false, afrag.v, false, bb, (short)0, acc2[nt], false, false);
        }
    }
    #pragma unroll
    for (int nt = 0; nt < 8; ++nt) {
        #pragma unroll
        for (int vi = 0; vi < 8; ++vi) {
            float x = acc2[nt][vi];
            size_t r = rowBase + waveRow + halfSel * 8 + vi;
            Hout[r * DIM + nt * 16 + lmod] = x > 0.f ? x : 0.f;
        }
    }
}

extern "C" void kernel_launch(void* const* d_in, const int* in_sizes, int n_in,
                              void* d_out, int out_size, void* d_ws, size_t ws_size,
                              hipStream_t stream) {
    (void)in_sizes; (void)n_in; (void)out_size; (void)ws_size;
    const float* A  = (const float*)d_in[0];   // A_norm: N x N
    const float* X  = (const float*)d_in[1];   // X:      N x 128
    const float* W0 = (const float*)d_in[2];   // 128 x 128
    const float* W1 = (const float*)d_in[3];   // 128 x 128
    float* out = (float*)d_out;                // N x 128
    float* H0  = (float*)d_ws;                 // N x 128 fp32 intermediate (8 MB)

    dim3 grid(NNODES / TILE_M);                // 128 blocks
    dim3 block(256);                           // 8 wave32s
    gcn_layer_kernel<<<grid, block, 0, stream>>>(A, X,  W0, H0);
    gcn_layer_kernel<<<grid, block, 0, stream>>>(A, H0, W1, out);
}
